// BurstAlign_36292473651319
// MI455X (gfx1250) — matrix operand order repaired and checked by reading, not compiled, not run on hardware
//
#include <hip/hip_runtime.h>
#include <hip/hip_bf16.h>

typedef __attribute__((ext_vector_type(16))) _Float16 v16h;
typedef __attribute__((ext_vector_type(8)))  float    v8f;

#define KC 192              // K-chunk, multiple of 32
#define RS (KC + 16)        // LDS row stride in f16 (416B -> 32B aligned rows)

// ---------------- scalar helper kernels ----------------

__global__ void k_cvt_f16(const float* __restrict__ s, _Float16* __restrict__ d, int n) {
  int i = blockIdx.x * blockDim.x + threadIdx.x;
  if (i < n) d[i] = (_Float16)s[i];
}

// build concat(ref, cur) for the offset-prediction branches
// feat: images of 64ch, out: 8 images of 128ch
__global__ void k_cat(const float* __restrict__ feat, float* __restrict__ out, int HW) {
  int idx = blockIdx.x * blockDim.x + threadIdx.x;
  int total = 8 * 128 * HW;
  if (idx >= total) return;
  int p  = idx % HW;
  int t  = idx / HW;
  int ch = t % 128;
  int i  = t / 128;
  int b = i >> 2, jj = i & 3;
  int frame = jj + (jj >= 2 ? 1 : 0);          // skip center frame 2
  int srcImg = (ch < 64) ? (b * 5 + 2) : (b * 5 + frame);
  out[idx] = feat[((size_t)srcImg * 64 + (ch & 63)) * HW + p];
}

// off = 2*bilinear_up2(p2[:, :144]) + p1[:, :144] ; msk = sigmoid(p1[:, 144:])
__global__ void k_offmask(const float* __restrict__ P1, const float* __restrict__ P2,
                          float* __restrict__ OFF, float* __restrict__ MSK) {
  int idx = blockIdx.x * blockDim.x + threadIdx.x;
  const int HW = 16384;
  int total = 8 * 216 * HW;
  if (idx >= total) return;
  int p  = idx % HW;
  int t  = idx / HW;
  int ch = t % 216;
  int i  = t / 216;
  float v1 = P1[idx];
  if (ch < 144) {
    int y = p >> 7, x = p & 127;
    float sy = 0.5f * (float)y - 0.25f;        // half-pixel centers, scale 2
    float sx = 0.5f * (float)x - 0.25f;
    sy = fminf(fmaxf(sy, 0.f), 63.f);
    sx = fminf(fmaxf(sx, 0.f), 63.f);
    int y0 = (int)sy, x0 = (int)sx;
    int y1 = min(y0 + 1, 63), x1 = min(x0 + 1, 63);
    float wy = sy - (float)y0, wx = sx - (float)x0;
    const float* src = P2 + ((size_t)i * 216 + ch) * 4096;
    float v = (1.f - wy) * ((1.f - wx) * src[y0 * 64 + x0] + wx * src[y0 * 64 + x1])
            +        wy  * ((1.f - wx) * src[y1 * 64 + x0] + wx * src[y1 * 64 + x1]);
    OFF[((size_t)i * 144 + ch) * HW + p] = 2.f * v + v1;
  } else {
    MSK[((size_t)i * 72 + (ch - 144)) * HW + p] = 1.f / (1.f + __expf(-v1));
  }
}

__global__ void k_refcopy(const float* __restrict__ f1, float* __restrict__ out) {
  int idx = blockIdx.x * blockDim.x + threadIdx.x;
  const int SL = 64 * 16384;
  if (idx >= 2 * SL) return;
  int b = idx / SL, e = idx % SL;
  size_t off = (size_t)(b * 5 + 2) * SL + e;
  out[off] = f1[off];
}

// ---------------- WMMA fragment helpers ----------------
// A (16x32 f16): lane L holds row M=L&15; half h=L>>4; VGPR v elem e:
//   K = 16h + 2v + e (16 contiguous)
// B (32x16 f16): lane L holds col N=L&15; half h: K = 16h + 2v + e (16 contiguous)
__device__ __forceinline__ v16h load_fragA(const _Float16* __restrict__ row, int h8) {
  v16h a;
#pragma unroll
  for (int vv = 0; vv < 8; ++vv) {
    int ka = 2 * vv + 2 * h8;
    a[2 * vv]     = row[ka];
    a[2 * vv + 1] = row[ka + 1];
  }
  return a;
}

// ---------------- implicit-GEMM 3x3 conv (pad 1) ----------------
// D[co][px] = sum_k W[co][k] * X[px][k],  k = cin*9 + ty*3 + tx
// block = 256 thr (8 waves): 64 out-channels x 64 pixels per workgroup;
// each wave owns one ctile and TWO pixel tiles (A fragment reused twice).
__global__ __launch_bounds__(256)
void k_conv3x3_wmma(const float* __restrict__ in, const _Float16* __restrict__ wf,
                    const float* __restrict__ bias, float* __restrict__ out,
                    int Cin, int Cout, int Hin, int Win, int Hout, int Wout,
                    int stride, int relu) {
  __shared__ _Float16 sX[64 * RS];
  __shared__ _Float16 sW[64 * RS];
  const int K   = Cin * 9;
  const int HWo = Hout * Wout;
  const int HWi = Hin * Win;
  const int img = blockIdx.z;
  const int px0 = blockIdx.x * 64;
  const int c0  = blockIdx.y * 64;
  const int tid = threadIdx.x;
  const int lane = tid & 31, wave = tid >> 5;
  const int ctile = wave & 3;          // 0..3  -> 16 channels each
  const int pg    = wave >> 2;         // 0..1  -> two pixel tiles each
  const float* inp = in + (size_t)img * Cin * HWi;

  v8f acc0 = {}, acc1 = {};
  for (int kc = 0; kc < K; kc += KC) {
    // stage im2col tile X[64][KC]
    for (int idx = tid; idx < 64 * KC; idx += 256) {
      int px = idx / KC, kk = idx - px * KC, k = kc + kk;
      float v = 0.f;
      if (k < K) {
        int cin = k / 9, tap = k - cin * 9;
        int p = px0 + px, oy = p / Wout, ox = p - oy * Wout;
        int iy = oy * stride + tap / 3 - 1;
        int ix = ox * stride + tap % 3 - 1;
        if (iy >= 0 && iy < Hin && ix >= 0 && ix < Win)
          v = inp[(size_t)cin * HWi + iy * Win + ix];
      }
      sX[px * RS + kk] = (_Float16)v;
    }
    // stage weights W[64][KC]
    for (int idx = tid; idx < 64 * KC; idx += 256) {
      int n = idx / KC, kk = idx - n * KC, k = kc + kk, c = c0 + n;
      _Float16 v = (_Float16)0.f;
      if (k < K && c < Cout) v = wf[(size_t)c * K + k];
      sW[n * RS + kk] = v;
    }
    __syncthreads();
    const _Float16* wrow  = &sW[(ctile * 16 + (lane & 15)) * RS];
    const _Float16* xrow0 = &sX[((pg * 2 + 0) * 16 + (lane & 15)) * RS];
    const _Float16* xrow1 = &sX[((pg * 2 + 1) * 16 + (lane & 15)) * RS];
    const int h8  = (lane >> 4) * 8;
    const int h16 = (lane >> 4) * 16;
#pragma unroll
    for (int kk = 0; kk < KC; kk += 32) {
      v16h a  = load_fragA(wrow + kk, h8);
      v16h b0 = *(const v16h*)(xrow0 + kk + h16);   // 32B-aligned (RS*2 = 416)
      v16h b1 = *(const v16h*)(xrow1 + kk + h16);
      acc0 = __builtin_amdgcn_wmma_f32_16x16x32_f16(false, a, false, b0,
                                                    (short)0, acc0, false, false);
      acc1 = __builtin_amdgcn_wmma_f32_16x16x32_f16(false, a, false, b1,
                                                    (short)0, acc1, false, false);
    }
    __syncthreads();
  }
  // D layout: VGPR i, lane L -> M = i + 8*(L>>4) (channel), N = L&15 (pixel)
  float* outp = out + (size_t)img * Cout * HWo;
  const int p0 = px0 + (pg * 2 + 0) * 16 + (lane & 15);
  const int p1 = px0 + (pg * 2 + 1) * 16 + (lane & 15);
#pragma unroll
  for (int i = 0; i < 8; ++i) {
    int co = c0 + ctile * 16 + i + ((lane >> 4) << 3);
    if (co < Cout) {
      float bv = bias[co];
      float v0 = acc0[i] + bv;
      float v1 = acc1[i] + bv;
      if (relu) { v0 = fmaxf(v0, 0.f); v1 = fmaxf(v1, 0.f); }
      outp[(size_t)co * HWo + p0] = v0;
      outp[(size_t)co * HWo + p1] = v1;
    }
  }
}

// ---------------- deformable conv: masked bilinear gather + WMMA GEMM ----------------
__global__ __launch_bounds__(256)
void k_deform_wmma(const float* __restrict__ feat,   // f1: 10 x 64 x 128 x 128
                   const float* __restrict__ OFF,    // 8 x 144 x HW
                   const float* __restrict__ MSK,    // 8 x 72 x HW
                   const _Float16* __restrict__ wf,  // 64 x 576
                   const float* __restrict__ bias,
                   float* __restrict__ out) {        // (2,5,64,128,128)
  __shared__ _Float16 sX[64 * RS];
  __shared__ _Float16 sW[64 * RS];
  const int K = 576, HW = 16384;
  const int i = blockIdx.z;                          // 0..7 non-center images
  const int b = i >> 2, jj = i & 3;
  const int frame = jj + (jj >= 2 ? 1 : 0);
  const float* fimg = feat + (size_t)(b * 5 + frame) * 64 * HW;
  const float* offp = OFF + (size_t)i * 144 * HW;
  const float* mskp = MSK + (size_t)i * 72 * HW;
  const int px0 = blockIdx.x * 64;
  const int tid = threadIdx.x, lane = tid & 31, wave = tid >> 5;
  const int ctile = wave & 3;
  const int pg    = wave >> 2;

  v8f acc0 = {}, acc1 = {};
  for (int kc = 0; kc < K; kc += KC) {               // 3 exact chunks
    for (int idx = tid; idx < 64 * KC; idx += 256) {
      int px = idx / KC, kk = idx - px * KC, k = kc + kk;
      int c = k / 9, tap = k - c * 9, g = c >> 3;    // Cg = 8
      int p = px0 + px, y = p >> 7, x = p & 127;
      int gk = g * 9 + tap;
      float oy = offp[(size_t)(gk * 2)     * HW + p];
      float ox = offp[(size_t)(gk * 2 + 1) * HW + p];
      float m  = mskp[(size_t)gk * HW + p];
      float sy = (float)y + (float)(tap / 3 - 1) + oy;
      float sx = (float)x + (float)(tap % 3 - 1) + ox;
      float y0f = floorf(sy), x0f = floorf(sx);
      float wy = sy - y0f, wx = sx - x0f;
      int y0 = (int)y0f, x0 = (int)x0f;
      const float* chp = fimg + (size_t)c * HW;
      float v = 0.f;
#pragma unroll
      for (int dy = 0; dy < 2; ++dy)
#pragma unroll
        for (int dx = 0; dx < 2; ++dx) {
          int yy = y0 + dy, xx = x0 + dx;
          if (yy >= 0 && yy < 128 && xx >= 0 && xx < 128) {
            float w = (dy ? wy : 1.f - wy) * (dx ? wx : 1.f - wx);
            v += chp[yy * 128 + xx] * w;
          }
        }
      sX[px * RS + kk] = (_Float16)(v * m);
    }
    for (int idx = tid; idx < 64 * KC; idx += 256) {
      int n = idx / KC, kk = idx - n * KC;
      sW[n * RS + kk] = wf[(size_t)n * K + kc + kk];
    }
    __syncthreads();
    const _Float16* wrow  = &sW[(ctile * 16 + (lane & 15)) * RS];
    const _Float16* xrow0 = &sX[((pg * 2 + 0) * 16 + (lane & 15)) * RS];
    const _Float16* xrow1 = &sX[((pg * 2 + 1) * 16 + (lane & 15)) * RS];
    const int h8  = (lane >> 4) * 8;
    const int h16 = (lane >> 4) * 16;
#pragma unroll
    for (int kk = 0; kk < KC; kk += 32) {
      v16h a  = load_fragA(wrow + kk, h8);
      v16h b0 = *(const v16h*)(xrow0 + kk + h16);
      v16h b1 = *(const v16h*)(xrow1 + kk + h16);
      acc0 = __builtin_amdgcn_wmma_f32_16x16x32_f16(false, a, false, b0,
                                                    (short)0, acc0, false, false);
      acc1 = __builtin_amdgcn_wmma_f32_16x16x32_f16(false, a, false, b1,
                                                    (short)0, acc1, false, false);
    }
    __syncthreads();
  }
  float* outp = out + (size_t)(b * 5 + frame) * 64 * HW;
  const int p0 = px0 + (pg * 2 + 0) * 16 + (lane & 15);
  const int p1 = px0 + (pg * 2 + 1) * 16 + (lane & 15);
#pragma unroll
  for (int ii = 0; ii < 8; ++ii) {
    int co = ctile * 16 + ii + ((lane >> 4) << 3);
    float bv = bias[co];
    outp[(size_t)co * HW + p0] = acc0[ii] + bv;
    outp[(size_t)co * HW + p1] = acc1[ii] + bv;
  }
}

// ---------------- host side ----------------

static inline void cvt(const float* s, _Float16* d, int n, hipStream_t st) {
  k_cvt_f16<<<(n + 255) / 256, 256, 0, st>>>(s, d, n);
}

extern "C" void kernel_launch(void* const* d_in, const int* in_sizes, int n_in,
                              void* d_out, int out_size, void* d_ws, size_t ws_size,
                              hipStream_t stream) {
  (void)in_sizes; (void)n_in; (void)out_size; (void)ws_size;
  const float* x       = (const float*)d_in[0];
  const float* fe1_w1  = (const float*)d_in[1];  const float* fe1_b1 = (const float*)d_in[2];
  const float* fe1_w2  = (const float*)d_in[3];  const float* fe1_b2 = (const float*)d_in[4];
  const float* fe1_w3  = (const float*)d_in[5];  const float* fe1_b3 = (const float*)d_in[6];
  const float* fe2_w1  = (const float*)d_in[7];  const float* fe2_b1 = (const float*)d_in[8];
  const float* fe2_w2  = (const float*)d_in[9];  const float* fe2_b2 = (const float*)d_in[10];
  const float* op1_w1  = (const float*)d_in[11]; const float* op1_b1 = (const float*)d_in[12];
  const float* op1_w2  = (const float*)d_in[13]; const float* op1_b2 = (const float*)d_in[14];
  const float* op1_w3  = (const float*)d_in[15]; const float* op1_b3 = (const float*)d_in[16];
  const float* op2_w1  = (const float*)d_in[17]; const float* op2_b1 = (const float*)d_in[18];
  const float* op2_w2  = (const float*)d_in[19]; const float* op2_b2 = (const float*)d_in[20];
  const float* op2_w3  = (const float*)d_in[21]; const float* op2_b3 = (const float*)d_in[22];
  const float* dcn_w   = (const float*)d_in[23]; const float* dcn_b  = (const float*)d_in[24];
  float* outp = (float*)d_out;

  // workspace arena (float units)
  float* ws = (float*)d_ws;
  size_t o = 0;
  float* A    = ws + o; o += 10485760ull;  // f1 (10x64x128x128), persists
  float* Bp   = ws + o; o += 10485760ull;  // big ping-pong
  float* Cp   = ws + o; o += 2621440ull;   // 64^2 ping
  float* Dp   = ws + o; o += 2621440ull;   // 64^2 pong / f2
  float* Ec   = ws + o; o += 16777216ull;  // cat1 (8x128x128x128), reused as op1 pong
  float* Fc   = ws + o; o += 4194304ull;   // cat2 (8x128x64x64)
  float* P2   = ws + o; o += 7077888ull;   // op2 head out (8x216x64x64)
  float* P1   = ws + o; o += 28311552ull;  // op1 head out (8x216x128x128); also fe1 scratch
  float* OFFb = ws + o; o += 18874368ull;  // 8x144x128x128
  float* MSKb = ws + o; o += 9437184ull;   // 8x72x128x128
  _Float16* W16 = (_Float16*)(ws + o);

  // f16 weight table
  _Float16* wfe1_1 = W16;            cvt(fe1_w1, wfe1_1, 2304,   stream);
  _Float16* wfe1_2 = wfe1_1 + 2304;  cvt(fe1_w2, wfe1_2, 36864,  stream);
  _Float16* wfe1_3 = wfe1_2 + 36864; cvt(fe1_w3, wfe1_3, 36864,  stream);
  _Float16* wfe2_1 = wfe1_3 + 36864; cvt(fe2_w1, wfe2_1, 36864,  stream);
  _Float16* wfe2_2 = wfe2_1 + 36864; cvt(fe2_w2, wfe2_2, 36864,  stream);
  _Float16* wop1_1 = wfe2_2 + 36864; cvt(op1_w1, wop1_1, 73728,  stream);
  _Float16* wop1_2 = wop1_1 + 73728; cvt(op1_w2, wop1_2, 36864,  stream);
  _Float16* wop1_3 = wop1_2 + 36864; cvt(op1_w3, wop1_3, 124416, stream);
  _Float16* wop2_1 = wop1_3 + 124416;cvt(op2_w1, wop2_1, 73728,  stream);
  _Float16* wop2_2 = wop2_1 + 73728; cvt(op2_w2, wop2_2, 36864,  stream);
  _Float16* wop2_3 = wop2_2 + 36864; cvt(op2_w3, wop2_3, 124416, stream);
  _Float16* wdcn   = wop2_3 + 124416;cvt(dcn_w,  wdcn,   36864,  stream);

  dim3 blk(256);
  // feature extraction L1 (10 images @128^2): x -> Bp -> P1 -> A  (f1 = A)
  k_conv3x3_wmma<<<dim3(256,1,10), blk, 0, stream>>>(x,  wfe1_1, fe1_b1, Bp, 4, 64,128,128,128,128,1,1);
  k_conv3x3_wmma<<<dim3(256,1,10), blk, 0, stream>>>(Bp, wfe1_2, fe1_b2, P1, 64,64,128,128,128,128,1,1);
  k_conv3x3_wmma<<<dim3(256,1,10), blk, 0, stream>>>(P1, wfe1_3, fe1_b3, A,  64,64,128,128,128,128,1,1);
  // feature extraction L2 (stride 2, then 3x3): A -> Cp -> Dp (f2 = Dp)
  k_conv3x3_wmma<<<dim3(64,1,10), blk, 0, stream>>>(A,  wfe2_1, fe2_b1, Cp, 64,64,128,128,64,64,2,1);
  k_conv3x3_wmma<<<dim3(64,1,10), blk, 0, stream>>>(Cp, wfe2_2, fe2_b2, Dp, 64,64,64,64,64,64,1,1);
  // concat(ref2, cur2) -> Fc
  k_cat<<<(8*128*4096 + 255)/256, blk, 0, stream>>>(Dp, Fc, 4096);
  // op2 head: Fc -> Cp -> Dp -> P2   (f2 dead after cat2)
  k_conv3x3_wmma<<<dim3(64,1,8), blk, 0, stream>>>(Fc, wop2_1, op2_b1, Cp, 128,64,64,64,64,64,1,1);
  k_conv3x3_wmma<<<dim3(64,1,8), blk, 0, stream>>>(Cp, wop2_2, op2_b2, Dp, 64, 64,64,64,64,64,1,1);
  k_conv3x3_wmma<<<dim3(64,4,8), blk, 0, stream>>>(Dp, wop2_3, op2_b3, P2, 64,216,64,64,64,64,1,0);
  // concat(ref1, cur1) -> Ec
  k_cat<<<(8*128*16384 + 255)/256, blk, 0, stream>>>(A, Ec, 16384);
  // op1 head: Ec -> Bp -> Ec -> P1
  k_conv3x3_wmma<<<dim3(256,1,8), blk, 0, stream>>>(Ec, wop1_1, op1_b1, Bp, 128,64,128,128,128,128,1,1);
  k_conv3x3_wmma<<<dim3(256,1,8), blk, 0, stream>>>(Bp, wop1_2, op1_b2, Ec, 64, 64,128,128,128,128,1,1);
  k_conv3x3_wmma<<<dim3(256,4,8), blk, 0, stream>>>(Ec, wop1_3, op1_b3, P1, 64,216,128,128,128,128,1,0);
  // fuse offsets (upsample x2 * 2 + p1) and masks (sigmoid)
  k_offmask<<<(8*216*16384 + 255)/256, blk, 0, stream>>>(P1, P2, OFFb, MSKb);
  // deformable conv -> aligned frames directly into d_out
  k_deform_wmma<<<dim3(256,1,8), blk, 0, stream>>>(A, OFFb, MSKb, wdcn, dcn_b, outp);
  // center reference frame passthrough
  k_refcopy<<<(2*64*16384 + 255)/256, blk, 0, stream>>>(A, outp);
}
